// LinearAttention_7172595384818
// MI455X (gfx1250) — matrix operand order repaired and checked by reading, not compiled, not run on hardware
//
#include <hip/hip_runtime.h>

// ---------------------------------------------------------------------------
// LinearAttention for MI455X (gfx1250, wave32, WMMA).
// Pipeline: cvt(x->bf16), transpose(Wqkv,Wout->bf16), WMMA bf16 GEMM (qkv),
// softmax_q, softmax_k + v scale, context (WMMA f32 16x16x4),
// out-einsum (WMMA f32 16x16x4, ctx staged via async-to-LDS),
// WMMA bf16 GEMM + bias + LayerNorm.
// Round 2: 2x2 register blocking in both bf16 GEMMs (halves VMEM per WMMA),
// GLOBAL_LOAD_ASYNC_TO_LDS_B128 + s_wait_asynccnt for ctx staging.
// ---------------------------------------------------------------------------

typedef __attribute__((ext_vector_type(16))) __bf16 v16bf;
typedef __attribute__((ext_vector_type(8)))  __bf16 v8bf;
typedef __attribute__((ext_vector_type(8)))  float  v8f;
typedef __attribute__((ext_vector_type(2)))  float  v2f;

#define BATCH 16
#define TOK   4096        // 64*64 tokens per batch
#define CH    256
#define F3    768
#define HEADS 8
#define DH    32
#define NROWS 65536       // BATCH*TOK

__device__ __forceinline__ unsigned short f2bf(float f) {
    unsigned u = __builtin_bit_cast(unsigned, f);
    u += 0x7fffu + ((u >> 16) & 1u);          // round-to-nearest-even
    return (unsigned short)(u >> 16);
}

__device__ __forceinline__ v16bf cat8(v8bf lo, v8bf hi) {
    return __builtin_shufflevector(lo, hi, 0,1,2,3,4,5,6,7,8,9,10,11,12,13,14,15);
}

// A fragment: 16 contiguous K values (matches B)
__device__ __forceinline__ v16bf loadA(const unsigned short* p) {
    return cat8(*(const v8bf*)p, *(const v8bf*)(p + 8));
}
// B fragment: rows striped across lanes -> 16 contiguous K values
__device__ __forceinline__ v16bf loadB(const unsigned short* p) {
    return cat8(*(const v8bf*)p, *(const v8bf*)(p + 8));
}

// ---------------------------------------------------------------- cvt x->bf16
__global__ __launch_bounds__(256) void k_cvt_x(const float* __restrict__ x,
                                               unsigned short* __restrict__ xb,
                                               int n) {
    int stride = gridDim.x * blockDim.x;
    for (int i = blockIdx.x * blockDim.x + threadIdx.x; i < n; i += stride)
        xb[i] = f2bf(x[i]);
}

// --------------------------------------------- transpose weight + cvt to bf16
// src: rows x cols (row-major fp32) -> dst: cols x rows (row-major bf16)
__global__ __launch_bounds__(256) void k_transpose_w(const float* __restrict__ src,
                                                     unsigned short* __restrict__ dst,
                                                     int rows, int cols) {
    int gid = blockIdx.x * blockDim.x + threadIdx.x;
    if (gid < rows * cols) {
        int r = gid / cols, c = gid % cols;
        dst[c * rows + r] = f2bf(src[gid]);
    }
}

// ----------------------------------------------------- GEMM1: qkv = x * Wqkv
// M=65536 K=256 N=768. Block: 8 waves in 2(M)x4(N) grid, each wave computes a
// 32x32 macro-tile as 2x2 WMMA tiles (4 frag loads -> 4 WMMAs per K-step).
__global__ __launch_bounds__(256) void k_gemm_qkv(const unsigned short* __restrict__ xb,
                                                  const unsigned short* __restrict__ wt,
                                                  float* __restrict__ qkv) {
    const int m0 = blockIdx.x * 64, n0 = blockIdx.y * 128;
    const int tid = threadIdx.x, w = tid >> 5, lane = tid & 31;
    const int wm = w >> 2, wn = w & 3;
    const int asel = (lane < 16) ? 0 : 16;
    const int bsel = (lane < 16) ? 0 : 16;
    const size_t ar0 = (size_t)(m0 + wm * 32 + (lane & 15)) * CH;
    const size_t ar1 = ar0 + (size_t)16 * CH;
    const size_t bc0 = (size_t)(n0 + wn * 32 + (lane & 15)) * CH;
    const size_t bc1 = bc0 + (size_t)16 * CH;
    v8f acc00 = {}, acc01 = {}, acc10 = {}, acc11 = {};
    for (int kk = 0; kk < CH; kk += 32) {
        v16bf a0 = loadA(xb + ar0 + kk + asel);
        v16bf a1 = loadA(xb + ar1 + kk + asel);
        v16bf b0 = loadB(wt + bc0 + kk + bsel);
        v16bf b1 = loadB(wt + bc1 + kk + bsel);
        acc00 = __builtin_amdgcn_wmma_f32_16x16x32_bf16(false, a0, false, b0, (short)0, acc00, false, false);
        acc01 = __builtin_amdgcn_wmma_f32_16x16x32_bf16(false, a0, false, b1, (short)0, acc01, false, false);
        acc10 = __builtin_amdgcn_wmma_f32_16x16x32_bf16(false, a1, false, b0, (short)0, acc10, false, false);
        acc11 = __builtin_amdgcn_wmma_f32_16x16x32_bf16(false, a1, false, b1, (short)0, acc11, false, false);
    }
    const int rbase = (lane < 16) ? 0 : 8;
    const int c0 = n0 + wn * 32 + (lane & 15), c1 = c0 + 16;
    #pragma unroll
    for (int r = 0; r < 8; ++r) {
        size_t r0 = (size_t)(m0 + wm * 32 + rbase + r) * F3;
        size_t r1 = r0 + (size_t)16 * F3;
        qkv[r0 + c0] = acc00[r];
        qkv[r0 + c1] = acc01[r];
        qkv[r1 + c0] = acc10[r];
        qkv[r1 + c1] = acc11[r];
    }
}

// ---------------------------------------------- softmax over d (32) + scale
// one thread per (b,h,n); writes qs[((b*8+h)*4096+n)*32 + d]
__global__ __launch_bounds__(256) void k_softmax_q(const float* __restrict__ qkv,
                                                   float* __restrict__ qs) {
    int gid = blockIdx.x * blockDim.x + threadIdx.x;   // 524288 total
    int b = gid >> 15, rem = gid & 32767;
    int hh = rem >> 12, n = rem & 4095;
    int c = ((n & 31) << 3) + hh;
    float v[32], mx = -1e30f;
    #pragma unroll
    for (int d = 0; d < 32; ++d) {
        float q = qkv[(size_t)((b << 12) + (d << 7) + (n >> 5)) * F3 + c];
        v[d] = q; mx = fmaxf(mx, q);
    }
    float s = 0.f;
    #pragma unroll
    for (int d = 0; d < 32; ++d) { v[d] = __expf(v[d] - mx); s += v[d]; }
    float inv = 0.17677669529663687f / s;              // (1/sqrt(32)) / sum
    size_t base = ((size_t)(((b << 3) + hh) << 12) + n) * 32;
    #pragma unroll
    for (int d = 0; d < 32; ++d) qs[base + d] = v[d] * inv;
}

// ------------------------- softmax over n (4096) for k ; v/4096 ; repack both
// one block per (b,h,d): ks/vs row of 4096, coalesced K-contiguous layout
__global__ __launch_bounds__(256) void k_softmax_kv(const float* __restrict__ qkv,
                                                    float* __restrict__ ks,
                                                    float* __restrict__ vs) {
    int d = blockIdx.x & 31, hh = (blockIdx.x >> 5) & 7, b = blockIdx.x >> 8;
    int tid = threadIdx.x;
    __shared__ float red[256];
    float kv[16], mx = -1e30f;
    #pragma unroll
    for (int j = 0; j < 16; ++j) {
        int n = tid + (j << 8);
        int t = (d << 7) + (n >> 5), c = ((n & 31) << 3) + hh;
        float val = qkv[(size_t)((b << 12) + t) * F3 + 256 + c];
        kv[j] = val; mx = fmaxf(mx, val);
    }
    red[tid] = mx; __syncthreads();
    for (int s = 128; s > 0; s >>= 1) {
        if (tid < s) red[tid] = fmaxf(red[tid], red[tid + s]);
        __syncthreads();
    }
    mx = red[0]; __syncthreads();
    float sum = 0.f;
    #pragma unroll
    for (int j = 0; j < 16; ++j) { kv[j] = __expf(kv[j] - mx); sum += kv[j]; }
    red[tid] = sum; __syncthreads();
    for (int s = 128; s > 0; s >>= 1) {
        if (tid < s) red[tid] += red[tid + s];
        __syncthreads();
    }
    float inv = 1.f / red[0];
    size_t rowoff = (size_t)blockIdx.x * 4096;   // blockIdx == (b*8+h)*32+d
    #pragma unroll
    for (int j = 0; j < 16; ++j) {
        int n = tid + (j << 8);
        ks[rowoff + n] = kv[j] * inv;
        int t = (d << 7) + (n >> 5), c = ((n & 31) << 3) + hh;
        vs[rowoff + n] = qkv[(size_t)((b << 12) + t) * F3 + 512 + c] * (1.f / 4096.f);
    }
}

// ---------------------- context[d][e] = sum_n ks[d][n]*vs[e][n] (per b,head)
// 1 block per (b,h); 8 waves: 4 output 16x16 tiles x 2 K-halves; f32 WMMA.
__global__ __launch_bounds__(256) void k_context(const float* __restrict__ ks,
                                                 const float* __restrict__ vs,
                                                 float* __restrict__ ctx) {
    int bh = blockIdx.x;
    int tid = threadIdx.x, w = tid >> 5, lane = tid & 31;
    int tile = w & 3, khalf = w >> 2;
    int dm = (tile >> 1) * 16, de = (tile & 1) * 16;
    int koff = (lane < 16) ? 0 : 2;
    size_t baseA = ((size_t)bh * 32 + dm + (lane & 15)) * 4096;
    size_t baseB = ((size_t)bh * 32 + de + (lane & 15)) * 4096;
    v8f acc = {};
    int k0 = khalf * 2048;
    for (int kb = k0; kb < k0 + 2048; kb += 4) {
        v2f a = *(const v2f*)(ks + baseA + kb + koff);
        v2f b = *(const v2f*)(vs + baseB + kb + koff);
        acc = __builtin_amdgcn_wmma_f32_16x16x4_f32(false, a, false, b,
                                                    (short)0, acc, false, false);
    }
    __shared__ float lds[4 * 256];
    int rbase = (lane < 16) ? 0 : 8;
    if (khalf == 1) {
        #pragma unroll
        for (int r = 0; r < 8; ++r)
            lds[tile * 256 + (rbase + r) * 16 + (lane & 15)] = acc[r];
    }
    __syncthreads();
    if (khalf == 0) {
        #pragma unroll
        for (int r = 0; r < 8; ++r) {
            float t = acc[r] + lds[tile * 256 + (rbase + r) * 16 + (lane & 15)];
            ctx[(size_t)bh * 1024 + (dm + rbase + r) * 32 + (de + (lane & 15))] = t;
        }
    }
}

// -------- out[e][n] = sum_d ctx[d][e]*qs[n][d]; scatter to attout bf16 (t,c)
// grid (64, 128): x = 64-col n-strip, y = (b*8+h).
// ctx (4KB) staged into LDS with GLOBAL_LOAD_ASYNC_TO_LDS_B128 (ASYNCcnt).
__global__ __launch_bounds__(256) void k_out(const float* __restrict__ ctx,
                                             const float* __restrict__ qs,
                                             unsigned short* __restrict__ ao) {
    int bh = blockIdx.y, b = bh >> 3, hh = bh & 7;
    int n0 = blockIdx.x * 64;
    int tid = threadIdx.x, w = tid >> 5, lane = tid & 31;
    __shared__ float c_s[1024];
    {
        // one b128 per lane: 256 lanes x 16B = whole 4KB ctx tile
        unsigned lds_addr = (unsigned)(unsigned long long)(const void*)&c_s[tid * 4];
        unsigned long long gaddr =
            (unsigned long long)(const void*)(ctx + (size_t)bh * 1024 + tid * 4);
        asm volatile("global_load_async_to_lds_b128 %0, %1, off"
                     :: "v"(lds_addr), "v"(gaddr) : "memory");
        asm volatile("s_wait_asynccnt 0" ::: "memory");
    }
    __syncthreads();
    int wm = w >> 2, wn = w & 3;
    int e_l = wm * 16 + (lane & 15);
    int ncol = n0 + wn * 16 + (lane & 15);
    int koff = (lane < 16) ? 0 : 2;
    size_t qbase = ((size_t)bh * 4096 + ncol) * 32;
    v8f acc = {};
    #pragma unroll
    for (int kb = 0; kb < 32; kb += 4) {
        v2f a; a.x = c_s[(kb + koff) * 32 + e_l];
        a.y = c_s[(kb + koff + 1) * 32 + e_l];            // A[e][k] = ctx[k][e]
        v2f bb = *(const v2f*)(qs + qbase + kb + koff);
        acc = __builtin_amdgcn_wmma_f32_16x16x4_f32(false, a, false, bb,
                                                    (short)0, acc, false, false);
    }
    int rbase = (lane < 16) ? 0 : 8;
    #pragma unroll
    for (int r = 0; r < 8; ++r) {
        int e = wm * 16 + rbase + r;
        int t = (e << 7) + (ncol >> 5);
        int c = ((ncol & 31) << 3) + hh;
        ao[(size_t)((b << 12) + t) * CH + c] = f2bf(acc[r]);
    }
}

// ------------- GEMM2 + bias + LayerNorm: y = LN(attout*Wout + bout), fp32 out
// 1 block per 32 token rows; 8 waves each own 32 of 256 cols; each wave does
// 2(M)x2(N) WMMA tiles -> A-fragment reuse x2, B-fragment reuse x2.
__global__ __launch_bounds__(256) void k_out_ln(const unsigned short* __restrict__ ao,
                                                const unsigned short* __restrict__ wt,
                                                const float* __restrict__ bout,
                                                const float* __restrict__ lns,
                                                const float* __restrict__ lnb,
                                                float* __restrict__ out) {
    const int row0 = blockIdx.x * 32;
    const int tid = threadIdx.x, w = tid >> 5, lane = tid & 31;
    const int asel = (lane < 16) ? 0 : 16;
    const int bsel = (lane < 16) ? 0 : 16;
    const size_t ar0 = (size_t)(row0 + (lane & 15)) * CH;
    const size_t ar1 = ar0 + (size_t)16 * CH;
    const int col0 = w * 32 + (lane & 15);
    const int col1 = col0 + 16;
    v8f acc00 = {}, acc01 = {}, acc10 = {}, acc11 = {};
    for (int kk = 0; kk < CH; kk += 32) {
        v16bf a0 = loadA(ao + ar0 + kk + asel);
        v16bf a1 = loadA(ao + ar1 + kk + asel);
        v16bf b0 = loadB(wt + (size_t)col0 * CH + kk + bsel);
        v16bf b1 = loadB(wt + (size_t)col1 * CH + kk + bsel);
        acc00 = __builtin_amdgcn_wmma_f32_16x16x32_bf16(false, a0, false, b0, (short)0, acc00, false, false);
        acc01 = __builtin_amdgcn_wmma_f32_16x16x32_bf16(false, a0, false, b1, (short)0, acc01, false, false);
        acc10 = __builtin_amdgcn_wmma_f32_16x16x32_bf16(false, a1, false, b0, (short)0, acc10, false, false);
        acc11 = __builtin_amdgcn_wmma_f32_16x16x32_bf16(false, a1, false, b1, (short)0, acc11, false, false);
    }
    __shared__ float tile[32 * 256];          // 32 KB
    __shared__ float mu[32], rs[32];
    const float bo0 = bout[col0], bo1 = bout[col1];
    const int rbase = (lane < 16) ? 0 : 8;
    #pragma unroll
    for (int r = 0; r < 8; ++r) {
        tile[(rbase + r) * 256 + col0]        = acc00[r] + bo0;
        tile[(rbase + r) * 256 + col1]        = acc01[r] + bo1;
        tile[(16 + rbase + r) * 256 + col0]   = acc10[r] + bo0;
        tile[(16 + rbase + r) * 256 + col1]   = acc11[r] + bo1;
    }
    __syncthreads();
    if (tid < 32) {
        float s = 0.f, s2 = 0.f;
        for (int c = 0; c < 256; ++c) {
            float xv = tile[tid * 256 + c];
            s += xv; s2 += xv * xv;
        }
        float m = s * (1.f / 256.f);
        mu[tid] = m;
        rs[tid] = rsqrtf(s2 * (1.f / 256.f) - m * m + 1e-6f);
    }
    __syncthreads();
    int r = tid >> 3, cb = (tid & 7) << 5;
    float m = mu[r], ri = rs[r];
    size_t obase = (size_t)(row0 + r) * 256;
    #pragma unroll
    for (int j = 0; j < 32; ++j) {
        int c = cb + j;
        out[obase + c] = (tile[r * 256 + c] - m) * ri * lns[c] + lnb[c];
    }
}

// ---------------------------------------------------------------------------
extern "C" void kernel_launch(void* const* d_in, const int* in_sizes, int n_in,
                              void* d_out, int out_size, void* d_ws, size_t ws_size,
                              hipStream_t stream) {
    const float* x    = (const float*)d_in[0];
    const float* Wqkv = (const float*)d_in[1];
    const float* Wout = (const float*)d_in[2];
    const float* bout = (const float*)d_in[3];
    const float* lns  = (const float*)d_in[4];
    const float* lnb  = (const float*)d_in[5];
    float* out = (float*)d_out;

    // workspace layout (bytes), ~449 MB total
    char* ws = (char*)d_ws;
    const size_t OFF_XB    = 0;                                  // 33,554,432
    const size_t OFF_WQKVT = OFF_XB    + (size_t)NROWS * CH * 2; //    393,216
    const size_t OFF_WOUTT = OFF_WQKVT + (size_t)F3 * CH * 2;    //    131,072
    const size_t OFF_QKV   = OFF_WOUTT + (size_t)CH * CH * 2;    // 201,326,592
    const size_t OFF_QS    = OFF_QKV   + (size_t)NROWS * F3 * 4; //  67,108,864
    const size_t OFF_KS    = OFF_QS    + (size_t)NROWS * CH * 4;
    const size_t OFF_VS    = OFF_KS    + (size_t)NROWS * CH * 4;
    const size_t OFF_CTX   = OFF_VS    + (size_t)NROWS * CH * 4; //     524,288
    const size_t OFF_AO    = OFF_CTX   + (size_t)128 * 1024 * 4; //  33,554,432

    unsigned short* xb     = (unsigned short*)(ws + OFF_XB);
    unsigned short* wqkvt  = (unsigned short*)(ws + OFF_WQKVT);
    unsigned short* woutt  = (unsigned short*)(ws + OFF_WOUTT);
    float*          qkv    = (float*)(ws + OFF_QKV);
    float*          qs     = (float*)(ws + OFF_QS);
    float*          ks     = (float*)(ws + OFF_KS);
    float*          vs     = (float*)(ws + OFF_VS);
    float*          ctx    = (float*)(ws + OFF_CTX);
    unsigned short* ao     = (unsigned short*)(ws + OFF_AO);

    k_cvt_x      <<<8192, 256, 0, stream>>>(x, xb, NROWS * CH);
    k_transpose_w<<<(CH * F3 + 255) / 256, 256, 0, stream>>>(Wqkv, wqkvt, CH, F3);
    k_transpose_w<<<(CH * CH + 255) / 256, 256, 0, stream>>>(Wout, woutt, CH, CH);
    k_gemm_qkv   <<<dim3(NROWS / 64, F3 / 128), 256, 0, stream>>>(xb, wqkvt, qkv);
    k_softmax_q  <<<2048, 256, 0, stream>>>(qkv, qs);
    k_softmax_kv <<<4096, 256, 0, stream>>>(qkv, ks, vs);
    k_context    <<<128, 256, 0, stream>>>(ks, vs, ctx);
    k_out        <<<dim3(64, 128), 256, 0, stream>>>(ctx, qs, ao);
    k_out_ln     <<<NROWS / 32, 256, 0, stream>>>(ao, woutt, bout, lns, lnb, out);
}
